// BiLstmCrf_44607530336615
// MI455X (gfx1250) — compile-verified, matrix operand order checked by reading, not run on hardware
//
#include <hip/hip_runtime.h>

// ---------------------------------------------------------------------------
// BiLSTM-CRF lattice model for MI455X (gfx1250, wave32, WMMA).
// Strategy: one persistent workgroup per (batch, direction) chain keeps the
// whole (S+1)xH h/c history in the 320KB WGP LDS; all GEMMs run on
// v_wmma_f32_16x16x32_bf16 with f32 accumulate; lattice matmuls are computed
// only for words that actually end at step t (exact, since masked words
// contribute zero). Weights are pre-transposed to bf16 once and stay L2-resident.
// ---------------------------------------------------------------------------

typedef unsigned short u16;
typedef __attribute__((ext_vector_type(16))) __bf16 v16bf;
typedef __attribute__((ext_vector_type(8)))  float  v8f;
typedef __attribute__((ext_vector_type(4)))  unsigned int u32x4;

#define TPB   384
#define WAVES 12

union Frag {
  v16bf v;
  u32x4 q[2];
  u16   s[16];
};

__device__ __forceinline__ u16 f2bf_u16(float f) {
  union { float f; unsigned int u; } c; c.f = f;
  unsigned int r = (c.u + 0x7FFFu + ((c.u >> 16) & 1u)) >> 16;
  return (u16)r;
}
__device__ __forceinline__ float sigf(float x) { return 1.0f / (1.0f + expf(-x)); }

__device__ __forceinline__ v8f wmma_bf16(const Frag& a, const Frag& b, v8f c) {
  return __builtin_amdgcn_wmma_f32_16x16x32_bf16(false, a.v, false, b.v, (short)0, c,
                                                 false, false);
}

// C(Mt*16 x Nt*16) = A(bf16, LDS, row-major, K=128) * B^T(bf16, global, [N][128])
// Output f32 with leading dim ldo, optional bias per column.
__device__ void run_gemm(const u16* __restrict__ A, const u16* __restrict__ BT,
                         float* __restrict__ OUT, int ldo, const float* __restrict__ bias,
                         int Mt, int Nt, int wv, int lane)
{
  const int khalf = (lane >> 4) << 3;   // 0 or 8 (K-half select per lane group)
  const int l15   = lane & 15;
  for (int tile = wv; tile < Mt * Nt; tile += WAVES) {
    const int mt = tile / Nt, nt = tile % Nt;
    const int m0 = mt * 16, n0 = nt * 16;
    v8f acc = {0.f,0.f,0.f,0.f,0.f,0.f,0.f,0.f};
    const u16* arow = A  + (m0 + l15) * 128 + khalf;
    const u16* brow = BT + (n0 + l15) * 128 + khalf;
#pragma unroll
    for (int kk = 0; kk < 128; kk += 32) {
      Frag fa, fb;
      fa.q[0] = *(const u32x4*)(arow + kk);
      fa.q[1] = *(const u32x4*)(arow + kk + 16);
      fb.q[0] = *(const u32x4*)(brow + kk);
      fb.q[1] = *(const u32x4*)(brow + kk + 16);
      acc = wmma_bf16(fa, fb, acc);
    }
    const int col = n0 + l15;
    const float bv = bias ? bias[col] : 0.0f;
#pragma unroll
    for (int r = 0; r < 8; ++r)
      OUT[(m0 + r + khalf) * ldo + col] = acc[r] + bv;
  }
}

// f32 [K][N] row-major  ->  bf16 [N][K] row-major (transposed, WMMA-friendly)
__global__ void wt_convert(const float* __restrict__ src, u16* __restrict__ dst,
                           int K, int N)
{
  int total = K * N;
  for (int e = blockIdx.x * blockDim.x + threadIdx.x; e < total;
       e += gridDim.x * blockDim.x) {
    int n = e / K, k = e - n * K;
    dst[e] = f2bf_u16(src[k * N + n]);
  }
}

struct ScanArgs {
  const int *char_ids, *kb_word_ids, *word_begin, *word_len, *seq_len;
  const float *char_emb, *kb_emb;
  const float *bc[2], *bw[2], *bl[2];
  const u16 *WcxT[2], *WwxT[2], *WlxT[2], *WchT[2], *WwhT[2], *WlcT[2];
  float *XgWs, *XlWs, *WgWs, *feats;
};

// Dynamic LDS layout (bytes): see carve below; total = 284,424 (< 320KB/WGP)
static constexpr size_t SMEM_BYTES =
    (size_t)(129 * 128 * 2 + 64 * 384 + 64 * 128 + 512 + 256) * sizeof(float) +
    (size_t)(64 * 128 + 128) * sizeof(u16) +
    (size_t)(129 + 129 + 64 + 64) * sizeof(int);

__global__ __launch_bounds__(TPB) void lattice_scan(ScanArgs A)
{
  extern __shared__ char smem[];
  float* hh    = (float*)smem;              // (S+1)*H history
  float* ch    = hh + 129 * 128;
  float* GW    = ch + 129 * 128;            // 64x384 per-step GEMM out (+ Xbf staging)
  float* CB    = GW + 64 * 384;             // 64x128 cb -> cw (in place)
  float* gates = CB + 64 * 128;             // 512 raw gate pre-activations
  float* red   = gates + 512;               // 128 sum_ew + 128 sum_ew*cw
  u16*  ABUF   = (u16*)(red + 256);         // 64x128 bf16 staging (Xw / hb / cw)
  u16*  hbf    = ABUF + 64 * 128;           // 128 bf16 h_prev
  int*  woff   = (int*)(hbf + 128);         // CSR offsets by end-position
  int*  wfill  = woff + 129;
  int*  worder = wfill + 129;
  int*  wbegA  = worder + 64;

  const int tid  = threadIdx.x;
  const int lane = tid & 31;
  const int wv   = tid >> 5;
  const int p    = blockIdx.x;              // (b, dir) chain
  const int b    = p >> 1;
  const int dir  = p & 1;
  const int Lb   = A.seq_len[b];

  const u16 *WcxT = A.WcxT[dir], *WwxT = A.WwxT[dir], *WlxT = A.WlxT[dir];
  const u16 *WchT = A.WchT[dir], *WwhT = A.WwhT[dir], *WlcT = A.WlcT[dir];
  const float *bc = A.bc[dir], *bw = A.bw[dir], *bl = A.bl[dir];
  float* XgP = A.XgWs + (size_t)p * 128 * 512;
  float* XlP = A.XlWs + (size_t)p * 128 * 128;
  float* WgP = A.WgWs + (size_t)p * 64 * 384;

  // zero h/c history (future slots must read as zero, matching the reference)
  for (int e = tid; e < 129 * 128; e += TPB) { hh[e] = 0.f; ch[e] = 0.f; }

  // CSR: valid words grouped by direction-adjusted end position
  if (tid == 0) {
    for (int i = 0; i <= 128; ++i) woff[i] = 0;
    for (int w = 0; w < 64; ++w) {
      int bg = A.word_begin[b * 64 + w], ln = A.word_len[b * 64 + w];
      int ef = bg + ln; if (ef > 127) ef = 127;
      if (ef < Lb) { int ed = dir ? (127 - bg) : ef; woff[ed + 1]++; }
    }
    for (int i = 1; i <= 128; ++i) woff[i] += woff[i - 1];
    for (int i = 0; i <= 128; ++i) wfill[i] = woff[i];
    for (int w = 0; w < 64; ++w) {
      int bg = A.word_begin[b * 64 + w], ln = A.word_len[b * 64 + w];
      int ef = bg + ln; if (ef > 127) ef = 127;
      if (ef < Lb) {
        int ed = dir ? (127 - bg) : ef;
        int bd = dir ? (127 - ef) : bg;
        int pos = wfill[ed]++;
        worder[pos] = w;
        wbegA[pos]  = bd;
      }
    }
  }

  // Prologue: embed gather (bf16) + the three time-parallel GEMMs
  u16* Xbf = (u16*)GW;                      // reuse GW region as staging
  for (int e = tid; e < 128 * 128; e += TPB) {
    int s = e >> 7, k = e & 127;
    int cid = A.char_ids[b * 128 + (dir ? 127 - s : s)];
    Xbf[e] = f2bf_u16(A.char_emb[cid * 128 + k]);
  }
  for (int e = tid; e < 64 * 128; e += TPB) {
    int w = e >> 7, k = e & 127;
    int wid = A.kb_word_ids[b * 64 + w];
    ABUF[e] = f2bf_u16(A.kb_emb[wid * 128 + k]);
  }
  __syncthreads();
  run_gemm(Xbf,  WcxT, XgP, 512, bc, 8, 32, wv, lane);   // Xg = X@Wcx + bc
  run_gemm(Xbf,  WlxT, XlP, 128, bl, 8,  8, wv, lane);   // Xl = X@Wlx + bl
  run_gemm(ABUF, WwxT, WgP, 384, bw, 4, 24, wv, lane);   // Wg = Xw@Wwx + bw
  __threadfence();
  __syncthreads();

  // Sequential lattice scan
  for (int t = 0; t < 128; ++t) {
    if (tid == 0 && t < 127) __builtin_prefetch(XgP + (t + 1) * 512, 0, 1);
    if (tid < 128) hbf[tid] = f2bf_u16(hh[t * 128 + tid]);
    __syncthreads();

    // gates = Xg[t] + h_prev @ Wch  (M=1 broadcast across the WMMA tile)
    {
      const int khalf = (lane >> 4) << 3;
      const u16* arow = hbf + khalf;        // all lanes read row 0
      for (int tile = wv; tile < 32; tile += WAVES) {
        int n0 = tile * 16;
        v8f acc = {0.f,0.f,0.f,0.f,0.f,0.f,0.f,0.f};
        const u16* brow = WchT + (n0 + (lane & 15)) * 128 + khalf;
#pragma unroll
        for (int kk = 0; kk < 128; kk += 32) {
          Frag fa, fb;
          fa.q[0] = *(const u32x4*)(arow + kk);
          fa.q[1] = *(const u32x4*)(arow + kk + 16);
          fb.q[0] = *(const u32x4*)(brow + kk);
          fb.q[1] = *(const u32x4*)(brow + kk + 16);
          acc = wmma_bf16(fa, fb, acc);
        }
        if (lane < 16) gates[n0 + lane] = acc[0] + XgP[t * 512 + n0 + lane];
      }
    }
    __syncthreads();

    const int nw = woff[t + 1] - woff[t];   // words ending exactly at t
    if (nw > 0) {
      const int base = woff[t];
      for (int e = tid; e < nw * 128; e += TPB) {
        int idx = e >> 7, j = e & 127;
        int bg = wbegA[base + idx];
        ABUF[idx * 128 + j] = f2bf_u16(hh[bg * 128 + j]);   // hb (bf16)
        CB[idx * 128 + j]   = ch[bg * 128 + j];             // cb (f32)
      }
      __syncthreads();
      const int Mt = (nw + 15) >> 4;
      run_gemm(ABUF, WwhT, GW, 384, nullptr, Mt, 24, wv, lane);  // hb@Wwh
      __syncthreads();
      for (int e = tid; e < nw * 128; e += TPB) {
        int idx = e >> 7, j = e & 127;
        int w = worder[base + idx];
        const float* wgr = WgP + w * 384;
        float wi = GW[idx * 384 + j]       + wgr[j];
        float wf = GW[idx * 384 + 128 + j] + wgr[128 + j];
        float wg = GW[idx * 384 + 256 + j] + wgr[256 + j];
        float cw = sigf(wf) * CB[idx * 128 + j] + sigf(wi) * tanhf(wg);
        CB[idx * 128 + j]   = cw;                  // cb buffer now holds cw
        ABUF[idx * 128 + j] = f2bf_u16(cw);        // bf16 copy for next GEMM
      }
      __syncthreads();
      run_gemm(ABUF, WlcT, GW, 384, nullptr, Mt, 8, wv, lane);   // cw@Wlc -> GW[:, :128]
      __syncthreads();
      if (tid < 128) {
        float se = 0.f, sec = 0.f;
        float xl = XlP[t * 128 + tid];
        for (int idx = 0; idx < nw; ++idx) {
          float lg = sigf(xl + GW[idx * 384 + tid]);
          float ew = expf(lg);
          se  += ew;
          sec += ew * CB[idx * 128 + tid];
        }
        red[tid] = se; red[128 + tid] = sec;
      }
      __syncthreads();
    }

    if (tid < 128) {
      const int j = tid;
      float iv = sigf(gates[j]);
      float fv = sigf(gates[128 + j]);
      float ov = sigf(gates[256 + j]);
      float gv = tanhf(gates[384 + j]);
      float hp = hh[t * 128 + j], cp = ch[t * 128 + j];
      float ct;
      if (nw > 0) {
        float ec = expf(iv);
        ct = (ec * gv + red[128 + j]) / (ec + red[j]);
      } else {
        ct = fv * cp + iv * gv;
      }
      float ht = ov * tanhf(ct);
      bool v = dir ? (t >= 128 - Lb) : (t < Lb);
      hh[(t + 1) * 128 + j] = v ? ht : hp;
      ch[(t + 1) * 128 + j] = v ? ct : cp;
      int so = dir ? (127 - t) : t;
      A.feats[((size_t)b * 128 + so) * 256 + dir * 128 + j] = v ? ht : 0.f;
    }
    __syncthreads();
  }
}

// logits[2048][32] = feats[2048][256] @ dense_W + b   (pure WMMA GEMM)
__global__ __launch_bounds__(256) void dense_logits_kernel(
    const float* __restrict__ feats, const u16* __restrict__ WT,
    const float* __restrict__ bias, float* __restrict__ logits)
{
  const int lane  = threadIdx.x & 31;
  const int wgl   = blockIdx.x * 8 + (threadIdx.x >> 5);   // 256 waves == 256 tiles
  const int mt = wgl >> 1, nt = wgl & 1;
  const int m0 = mt * 16, n0 = nt * 16;
  const int khalf = (lane >> 4) << 3;
  const int l15   = lane & 15;
  v8f acc = {0.f,0.f,0.f,0.f,0.f,0.f,0.f,0.f};
  const float* arow = feats + (m0 + l15) * 256;
  const u16*   brow = WT + (n0 + l15) * 256 + khalf;
#pragma unroll
  for (int kk = 0; kk < 256; kk += 32) {
    Frag fa, fb;
#pragma unroll
    for (int e = 0; e < 8; ++e) {
      fa.s[e]     = f2bf_u16(arow[kk + khalf + e]);
      fa.s[8 + e] = f2bf_u16(arow[kk + 16 + khalf + e]);
    }
    fb.q[0] = *(const u32x4*)(brow + kk);
    fb.q[1] = *(const u32x4*)(brow + kk + 16);
    acc = wmma_bf16(fa, fb, acc);
  }
  const int col = n0 + l15;
  const float bv = bias[col];
#pragma unroll
  for (int r = 0; r < 8; ++r)
    logits[(m0 + r + khalf) * 32 + col] = acc[r] + bv;
}

// CRF forward: L=32 maps exactly onto one wave32; one wave per batch element.
__global__ __launch_bounds__(512) void crf_kernel(
    const float* __restrict__ logits, const int* __restrict__ label,
    const int* __restrict__ seqlen, const float* __restrict__ T,
    float* __restrict__ out)
{
  __shared__ float Tl[1024];
  __shared__ float lossArr[16];
  const int tid = threadIdx.x;
  for (int e = tid; e < 1024; e += 512) Tl[e] = T[e];
  __syncthreads();
  const int b = tid >> 5, j = tid & 31;
  const int Lb = seqlen[b];
  const float* lg = logits + (size_t)b * 128 * 32;
  const int* lab = label + b * 128;

  float gp = 0.f;                                  // gold score
  for (int t = j; t < 128; t += 32)
    if (t < Lb) gp += lg[t * 32 + lab[t]];
  for (int t = j; t < 127; t += 32)
    if (t + 1 < Lb) gp += Tl[lab[t] * 32 + lab[t + 1]];
  for (int off = 16; off > 0; off >>= 1) gp += __shfl_xor(gp, off);

  float alpha = lg[j];                             // forward recursion
  for (int t = 1; t < 128; ++t) {
    if (t < Lb) {                                  // Lb is wave-uniform
      float mx = -3.4e38f;
      for (int i = 0; i < 32; ++i) {
        float ai = __shfl(alpha, i);
        mx = fmaxf(mx, ai + Tl[i * 32 + j]);
      }
      float s = 0.f;
      for (int i = 0; i < 32; ++i) {
        float ai = __shfl(alpha, i);
        s += expf(ai + Tl[i * 32 + j] - mx);
      }
      alpha = logf(s) + mx + lg[t * 32 + j];
    }
  }
  float mx = alpha;
  for (int off = 16; off > 0; off >>= 1) mx = fmaxf(mx, __shfl_xor(mx, off));
  float s = expf(alpha - mx);
  for (int off = 16; off > 0; off >>= 1) s += __shfl_xor(s, off);
  if (j == 0) lossArr[b] = (logf(s) + mx) - gp;
  __syncthreads();
  if (tid == 0) {
    float acc = 0.f;
    for (int i = 0; i < 16; ++i) acc += lossArr[i];
    out[0] = acc * (1.0f / 16.0f);
  }
}

extern "C" void kernel_launch(void* const* d_in, const int* in_sizes, int n_in,
                              void* d_out, int out_size, void* d_ws, size_t ws_size,
                              hipStream_t stream)
{
  (void)in_sizes; (void)n_in; (void)out_size; (void)ws_size;
  const int* char_ids    = (const int*)d_in[0];
  const int* kb_word_ids = (const int*)d_in[1];
  const int* word_begin  = (const int*)d_in[2];
  const int* word_len    = (const int*)d_in[3];
  const int* label       = (const int*)d_in[4];
  const int* seqlen      = (const int*)d_in[5];
  const float* char_emb  = (const float*)d_in[6];
  const float* kb_emb    = (const float*)d_in[7];
  const float* dense_W   = (const float*)d_in[8];
  const float* dense_b   = (const float*)d_in[9];
  const float* crf_T     = (const float*)d_in[10];
  const float* Wcx[2] = {(const float*)d_in[11], (const float*)d_in[20]};
  const float* Wch[2] = {(const float*)d_in[12], (const float*)d_in[21]};
  const float* bcv[2] = {(const float*)d_in[13], (const float*)d_in[22]};
  const float* Wwx[2] = {(const float*)d_in[14], (const float*)d_in[23]};
  const float* Wwh[2] = {(const float*)d_in[15], (const float*)d_in[24]};
  const float* bwv[2] = {(const float*)d_in[16], (const float*)d_in[25]};
  const float* Wlx[2] = {(const float*)d_in[17], (const float*)d_in[26]};
  const float* Wlc[2] = {(const float*)d_in[18], (const float*)d_in[27]};
  const float* blv[2] = {(const float*)d_in[19], (const float*)d_in[28]};

  char* ws = (char*)d_ws;
  size_t off = 0;
  auto carve = [&](size_t bytes) -> void* {
    void* r = ws + off;
    off = (off + bytes + 255) & ~(size_t)255;
    return r;
  };
  u16 *WcxT[2], *WwxT[2], *WlxT[2], *WchT[2], *WwhT[2], *WlcT[2];
  for (int d = 0; d < 2; ++d) {
    WcxT[d] = (u16*)carve(512 * 128 * 2);
    WwxT[d] = (u16*)carve(384 * 128 * 2);
    WlxT[d] = (u16*)carve(128 * 128 * 2);
    WchT[d] = (u16*)carve(512 * 128 * 2);
    WwhT[d] = (u16*)carve(384 * 128 * 2);
    WlcT[d] = (u16*)carve(128 * 128 * 2);
  }
  u16* denseWT  = (u16*)carve(32 * 256 * 2);
  float* XgWs   = (float*)carve((size_t)32 * 128 * 512 * 4);
  float* XlWs   = (float*)carve((size_t)32 * 128 * 128 * 4);
  float* WgWs   = (float*)carve((size_t)32 * 64 * 384 * 4);
  float* feats  = (float*)carve((size_t)16 * 128 * 256 * 4);
  float* logits = (float*)carve((size_t)16 * 128 * 32 * 4);

  for (int d = 0; d < 2; ++d) {
    wt_convert<<<64, 256, 0, stream>>>(Wcx[d], WcxT[d], 128, 512);
    wt_convert<<<64, 256, 0, stream>>>(Wwx[d], WwxT[d], 128, 384);
    wt_convert<<<64, 256, 0, stream>>>(Wlx[d], WlxT[d], 128, 128);
    wt_convert<<<64, 256, 0, stream>>>(Wch[d], WchT[d], 128, 512);
    wt_convert<<<64, 256, 0, stream>>>(Wwh[d], WwhT[d], 128, 384);
    wt_convert<<<64, 256, 0, stream>>>(Wlc[d], WlcT[d], 128, 128);
  }
  wt_convert<<<8, 256, 0, stream>>>(dense_W, denseWT, 256, 32);

  ScanArgs A;
  A.char_ids = char_ids; A.kb_word_ids = kb_word_ids;
  A.word_begin = word_begin; A.word_len = word_len; A.seq_len = seqlen;
  A.char_emb = char_emb; A.kb_emb = kb_emb;
  for (int d = 0; d < 2; ++d) {
    A.bc[d] = bcv[d]; A.bw[d] = bwv[d]; A.bl[d] = blv[d];
    A.WcxT[d] = WcxT[d]; A.WwxT[d] = WwxT[d]; A.WlxT[d] = WlxT[d];
    A.WchT[d] = WchT[d]; A.WwhT[d] = WwhT[d]; A.WlcT[d] = WlcT[d];
  }
  A.XgWs = XgWs; A.XlWs = XlWs; A.WgWs = WgWs; A.feats = feats;

  (void)hipFuncSetAttribute((const void*)lattice_scan,
                            hipFuncAttributeMaxDynamicSharedMemorySize,
                            (int)SMEM_BYTES);
  lattice_scan<<<32, TPB, SMEM_BYTES, stream>>>(A);
  dense_logits_kernel<<<32, 256, 0, stream>>>(feats, denseWT, dense_b, logits);
  crf_kernel<<<1, 512, 0, stream>>>(logits, label, seqlen, crf_T, (float*)d_out);
}
